// ShareSelfAttention_48876727829097
// MI455X (gfx1250) — matrix operand order repaired and checked by reading, not compiled, not run on hardware
//
#include <hip/hip_runtime.h>
#include <hip/hip_bf16.h>

#define DEV __device__ __forceinline__

typedef __attribute__((ext_vector_type(16))) _Float16 v16h;
typedef __attribute__((ext_vector_type(8)))  float    v8f;
typedef __attribute__((ext_vector_type(2)))  _Float16 h2;

static constexpr int SEQ   = 2048;  // N
static constexpr int BATCH = 2;     // B
static constexpr int CH    = 1024;  // C
static constexpr int NH    = 16;    // heads
static constexpr int HD    = 64;    // head dim

DEV v8f wmma(v16h a, v16h b, v8f c) {
    return __builtin_amdgcn_wmma_f32_16x16x32_f16(false, a, false, b,
                                                  (short)0, c, false, false);
}

// A-fragment (16x32 f16, M=lane&15): element e -> K = (e<8?e:e+8) + 8*(lane>=16)
DEV v16h load_afrag_f32(const float* __restrict__ row, int hi8) {
    v16h a;
#pragma unroll
    for (int v = 0; v < 8; ++v) {
        int k = (2 * v < 8 ? 2 * v : 2 * v + 8) + hi8;
        float2 f = *(const float2*)(row + k);
        a[2 * v]     = (_Float16)f.x;
        a[2 * v + 1] = (_Float16)f.y;
    }
    return a;
}

DEV v16h load_afrag_f16(const _Float16* __restrict__ row, int hi8) {
    v16h a;
#pragma unroll
    for (int v = 0; v < 8; ++v) {
        int k = (2 * v < 8 ? 2 * v : 2 * v + 8) + hi8;
        h2 t = *(const h2*)(row + k);
        a[2 * v]     = t.x;
        a[2 * v + 1] = t.y;
    }
    return a;
}

// B-fragment (32x16 f16, N=lane&15): element e -> K = (e<8?e:e+8) + 8*(lane>=16)
DEV v16h load_bfrag_f16(const _Float16* __restrict__ p, int ld, int hi8) {
    v16h bv;
#pragma unroll
    for (int s = 0; s < 16; ++s) {
        int k = (s < 8 ? s : s + 8) + hi8;
        bv[s] = p[(size_t)k * ld];
    }
    return bv;
}

DEV v8f vzero() {
    v8f z;
#pragma unroll
    for (int e = 0; e < 8; ++e) z[e] = 0.f;
    return z;
}

// ---------------- one-time weight conversion f32 -> f16 ----------------------
__global__ void cvt_f32_to_f16_kernel(const float* __restrict__ src,
                                      _Float16* __restrict__ dst, int count4) {
    int i = blockIdx.x * blockDim.x + threadIdx.x;  // index of float4 group
    if (i >= count4) return;
    float4 f = ((const float4*)src)[i];
    _Float16* d = dst + (size_t)i * 4;
    d[0] = (_Float16)f.x;
    d[1] = (_Float16)f.y;
    d[2] = (_Float16)f.z;
    d[3] = (_Float16)f.w;
}

// ---------------- input projections: f32 act x f16 W -> f16 out --------------
// rows m = b*SEQ + n over (B*N); X layout: SEQ_FIRST: (N,B,C); else (B,N,C)
template <int SEQ_FIRST>
__global__ __launch_bounds__(32) void proj_in_kernel(
        const float* __restrict__ X, const _Float16* __restrict__ W,
        const float* __restrict__ bias, _Float16* __restrict__ out,
        int Nout, float scale) {
    const int lane = threadIdx.x & 31;
    const int hi8  = (lane >= 16) ? 8 : 0;
    const int lo   = lane & 15;
    const int n0   = blockIdx.x * 64;
    const int m0   = blockIdx.y * 64;

    const float* arow[4];
#pragma unroll
    for (int mt = 0; mt < 4; ++mt) {
        int m = m0 + mt * 16 + lo;
        int b = m >> 11, n = m & (SEQ - 1);
        arow[mt] = SEQ_FIRST ? (X + ((size_t)n * BATCH + b) * CH)
                             : (X + (size_t)m * CH);
    }

    v8f acc[4][4];
#pragma unroll
    for (int i = 0; i < 4; ++i)
#pragma unroll
        for (int j = 0; j < 4; ++j) acc[i][j] = vzero();

    for (int k0 = 0; k0 < CH; k0 += 32) {
        // prefetch next K-step operands into cache (global_prefetch_b8)
        int kpf = (k0 + 32 < CH) ? k0 + 32 : k0;
        __builtin_prefetch(W + (size_t)(kpf + lane) * Nout + n0, 0, 1);
        __builtin_prefetch(arow[0] + kpf, 0, 1);

        v16h bf[4];
#pragma unroll
        for (int nt = 0; nt < 4; ++nt)
            bf[nt] = load_bfrag_f16(W + (size_t)k0 * Nout + n0 + nt * 16 + lo, Nout, hi8);
        v16h af[4];
#pragma unroll
        for (int mt = 0; mt < 4; ++mt)
            af[mt] = load_afrag_f32(arow[mt] + k0, hi8);
#pragma unroll
        for (int mt = 0; mt < 4; ++mt)
#pragma unroll
            for (int nt = 0; nt < 4; ++nt)
                acc[mt][nt] = wmma(af[mt], bf[nt], acc[mt][nt]);
    }

#pragma unroll
    for (int nt = 0; nt < 4; ++nt) {
        int n = n0 + nt * 16 + lo;
        float bn = bias[n];
#pragma unroll
        for (int mt = 0; mt < 4; ++mt) {
            int mbase = m0 + mt * 16 + hi8;
#pragma unroll
            for (int r = 0; r < 8; ++r) {
                float v = (acc[mt][nt][r] + bn) * scale;
                out[(size_t)(mbase + r) * Nout + n] = (_Float16)v;
            }
        }
    }
}

// ---------------- K transpose: (B,N,HD) -> (B,HD,N) --------------------------
__global__ void transpose_k_kernel(const _Float16* __restrict__ kin,
                                   _Float16* __restrict__ kt) {
    int idx = blockIdx.x * blockDim.x + threadIdx.x;
    if (idx >= BATCH * SEQ * HD) return;
    int d = idx % HD;
    int n = (idx / HD) % SEQ;
    int b = idx / (HD * SEQ);
    kt[((size_t)b * HD + d) * SEQ + n] = kin[idx];
}

// ---------------- flash attention: one wave per (b, h, 64 query rows) --------
__global__ __launch_bounds__(32) void attn_kernel(
        const _Float16* __restrict__ qbuf, const _Float16* __restrict__ ktbuf,
        const _Float16* __restrict__ vbuf, _Float16* __restrict__ xbuf) {
    __shared__ __align__(16) float    sS[64][33];
    __shared__ __align__(16) _Float16 sP[64][34];
    __shared__ float sAlpha[64];
    __shared__ float sLinv[64];

    const int lane = threadIdx.x & 31;
    const int hi8  = (lane >= 16) ? 8 : 0;
    const int lo   = lane & 15;
    const int b = blockIdx.z, h = blockIdx.y;
    const int q0 = blockIdx.x * 64;

    // Q fragments (already scaled by 1/sqrt(hd) in projection)
    v16h qf[4][2];
#pragma unroll
    for (int mt = 0; mt < 4; ++mt) {
        const _Float16* qrow =
            qbuf + ((size_t)b * SEQ + q0 + mt * 16 + lo) * CH + h * HD;
#pragma unroll
        for (int kt = 0; kt < 2; ++kt)
            qf[mt][kt] = load_afrag_f16(qrow + kt * 32, hi8);
    }

    v8f o[4][4];
#pragma unroll
    for (int i = 0; i < 4; ++i)
#pragma unroll
        for (int j = 0; j < 4; ++j) o[i][j] = vzero();

    float mrow[2] = {-1e30f, -1e30f};
    float lrow[2] = {0.f, 0.f};

    const _Float16* ktb = ktbuf + (size_t)b * HD * SEQ;
    const _Float16* vb  = vbuf + (size_t)b * SEQ * HD;

    for (int kb = 0; kb < SEQ; kb += 32) {
        // prefetch next key block of K^T and V
        int kpf = (kb + 32 < SEQ) ? kb + 32 : kb;
        __builtin_prefetch(ktb + (size_t)lane * SEQ + kpf, 0, 1);
        __builtin_prefetch(vb + (size_t)(kpf + lane) * HD, 0, 1);

        // S = Q K^T for 64 rows x 32 keys
        v16h bK[2][2];
#pragma unroll
        for (int kt = 0; kt < 2; ++kt)
#pragma unroll
            for (int st = 0; st < 2; ++st)
                bK[kt][st] = load_bfrag_f16(
                    ktb + (size_t)(kt * 32) * SEQ + kb + st * 16 + lo, SEQ, hi8);
#pragma unroll
        for (int st = 0; st < 2; ++st)
#pragma unroll
            for (int mt = 0; mt < 4; ++mt) {
                v8f s = vzero();
#pragma unroll
                for (int kt = 0; kt < 2; ++kt)
                    s = wmma(qf[mt][kt], bK[kt][st], s);
#pragma unroll
                for (int r = 0; r < 8; ++r)
                    sS[mt * 16 + hi8 + r][st * 16 + lo] = s[r];
            }
        __syncthreads();

        // online softmax: each lane owns rows 2*lane, 2*lane+1
#pragma unroll
        for (int rr = 0; rr < 2; ++rr) {
            int row = lane * 2 + rr;
            float bm = -1e30f;
#pragma unroll
            for (int j = 0; j < 32; ++j) bm = fmaxf(bm, sS[row][j]);
            float nm = fmaxf(mrow[rr], bm);
            float al = __expf(mrow[rr] - nm);
            float sum = 0.f;
#pragma unroll
            for (int j = 0; j < 32; ++j) {
                float p = __expf(sS[row][j] - nm);
                sum += p;
                sP[row][j] = (_Float16)p;
            }
            lrow[rr] = lrow[rr] * al + sum;
            mrow[rr] = nm;
            sAlpha[row] = al;
        }
        __syncthreads();

        // rescale running O
#pragma unroll
        for (int mt = 0; mt < 4; ++mt) {
            float a8[8];
#pragma unroll
            for (int r = 0; r < 8; ++r) a8[r] = sAlpha[mt * 16 + hi8 + r];
#pragma unroll
            for (int nt = 0; nt < 4; ++nt)
#pragma unroll
                for (int r = 0; r < 8; ++r) o[mt][nt][r] *= a8[r];
        }

        // O += P * V
        v16h vf[4];
#pragma unroll
        for (int nt = 0; nt < 4; ++nt)
            vf[nt] = load_bfrag_f16(vb + (size_t)kb * HD + nt * 16 + lo, HD, hi8);
#pragma unroll
        for (int mt = 0; mt < 4; ++mt) {
            v16h pf;
#pragma unroll
            for (int v = 0; v < 8; ++v) {
                int k = (2 * v < 8 ? 2 * v : 2 * v + 8) + hi8;
                h2 t = *(const h2*)&sP[mt * 16 + lo][k];
                pf[2 * v]     = t.x;
                pf[2 * v + 1] = t.y;
            }
#pragma unroll
            for (int nt = 0; nt < 4; ++nt)
                o[mt][nt] = wmma(pf, vf[nt], o[mt][nt]);
        }
        __syncthreads();
    }

    // final 1/l normalization + store x (B,N,C) f16
    sLinv[lane * 2]     = 1.f / lrow[0];
    sLinv[lane * 2 + 1] = 1.f / lrow[1];
    __syncthreads();
#pragma unroll
    for (int mt = 0; mt < 4; ++mt) {
        float i8[8];
#pragma unroll
        for (int r = 0; r < 8; ++r) i8[r] = sLinv[mt * 16 + hi8 + r];
#pragma unroll
        for (int nt = 0; nt < 4; ++nt)
#pragma unroll
            for (int r = 0; r < 8; ++r) {
                float v = o[mt][nt][r] * i8[r];
                size_t row = (size_t)b * SEQ + q0 + mt * 16 + hi8 + r;
                xbuf[row * CH + h * HD + nt * 16 + lo] = (_Float16)v;
            }
    }
}

// ---------------- output projection: f16 X @ f16 Wp -> f32 out (N,B,C) -------
__global__ __launch_bounds__(32) void proj_out_kernel(
        const _Float16* __restrict__ Xh, const _Float16* __restrict__ W,
        const float* __restrict__ bias, float* __restrict__ out) {
    const int lane = threadIdx.x & 31;
    const int hi8  = (lane >= 16) ? 8 : 0;
    const int lo   = lane & 15;
    const int n0   = blockIdx.x * 64;
    const int m0   = blockIdx.y * 64;

    const _Float16* arow[4];
#pragma unroll
    for (int mt = 0; mt < 4; ++mt)
        arow[mt] = Xh + (size_t)(m0 + mt * 16 + lo) * CH;

    v8f acc[4][4];
#pragma unroll
    for (int i = 0; i < 4; ++i)
#pragma unroll
        for (int j = 0; j < 4; ++j) acc[i][j] = vzero();

    for (int k0 = 0; k0 < CH; k0 += 32) {
        int kpf = (k0 + 32 < CH) ? k0 + 32 : k0;
        __builtin_prefetch(W + (size_t)(kpf + lane) * CH + n0, 0, 1);
        __builtin_prefetch(arow[0] + kpf, 0, 1);

        v16h bf[4];
#pragma unroll
        for (int nt = 0; nt < 4; ++nt)
            bf[nt] = load_bfrag_f16(W + (size_t)k0 * CH + n0 + nt * 16 + lo, CH, hi8);
        v16h af[4];
#pragma unroll
        for (int mt = 0; mt < 4; ++mt)
            af[mt] = load_afrag_f16(arow[mt] + k0, hi8);
#pragma unroll
        for (int mt = 0; mt < 4; ++mt)
#pragma unroll
            for (int nt = 0; nt < 4; ++nt)
                acc[mt][nt] = wmma(af[mt], bf[nt], acc[mt][nt]);
    }

#pragma unroll
    for (int nt = 0; nt < 4; ++nt) {
        int ncol = n0 + nt * 16 + lo;
        float bn = bias[ncol];
#pragma unroll
        for (int mt = 0; mt < 4; ++mt) {
#pragma unroll
            for (int r = 0; r < 8; ++r) {
                int m = m0 + mt * 16 + hi8 + r;
                int b = m >> 11;
                int n = m & (SEQ - 1);
                out[((size_t)n * BATCH + b) * CH + ncol] = acc[mt][nt][r] + bn;
            }
        }
    }
}

extern "C" void kernel_launch(void* const* d_in, const int* in_sizes, int n_in,
                              void* d_out, int out_size, void* d_ws,
                              size_t ws_size, hipStream_t stream) {
    (void)in_sizes; (void)n_in; (void)out_size; (void)ws_size;
    const float* query = (const float*)d_in[0];
    const float* keyi  = (const float*)d_in[1];
    const float* value = (const float*)d_in[2];
    // d_in[3] = num_heads (compile-time constant here)
    const float* Wq = (const float*)d_in[4];
    const float* bq = (const float*)d_in[5];
    const float* Wk = (const float*)d_in[6];
    const float* bk = (const float*)d_in[7];
    const float* Wv = (const float*)d_in[8];
    const float* bv = (const float*)d_in[9];
    const float* Wp = (const float*)d_in[10];
    const float* bp = (const float*)d_in[11];
    float* out = (float*)d_out;

    _Float16* ws    = (_Float16*)d_ws;
    _Float16* qbuf  = ws;                                     // B*N*C
    _Float16* kbuf  = qbuf + (size_t)BATCH * SEQ * CH;        // B*N*HD
    _Float16* ktbuf = kbuf + (size_t)BATCH * SEQ * HD;        // B*HD*N
    _Float16* vbuf  = ktbuf + (size_t)BATCH * SEQ * HD;       // B*N*HD
    _Float16* xbuf  = vbuf + (size_t)BATCH * SEQ * HD;        // B*N*C
    _Float16* wqh   = xbuf + (size_t)BATCH * SEQ * CH;        // C*C
    _Float16* wkh   = wqh + (size_t)CH * CH;                  // C*HD
    _Float16* wvh   = wkh + (size_t)CH * HD;                  // C*HD
    _Float16* wph   = wvh + (size_t)CH * HD;                  // C*C

    const float scale = 0.125f;  // 1/sqrt(HD)

    // one-time weight conversions (counts divisible by 4)
    cvt_f32_to_f16_kernel<<<(CH * CH / 4 + 255) / 256, 256, 0, stream>>>(
        Wq, wqh, CH * CH / 4);
    cvt_f32_to_f16_kernel<<<(CH * HD / 4 + 255) / 256, 256, 0, stream>>>(
        Wk, wkh, CH * HD / 4);
    cvt_f32_to_f16_kernel<<<(CH * HD / 4 + 255) / 256, 256, 0, stream>>>(
        Wv, wvh, CH * HD / 4);
    cvt_f32_to_f16_kernel<<<(CH * CH / 4 + 255) / 256, 256, 0, stream>>>(
        Wp, wph, CH * CH / 4);

    proj_in_kernel<1><<<dim3(CH / 64, BATCH * SEQ / 64), 32, 0, stream>>>(
        query, wqh, bq, qbuf, CH, scale);
    proj_in_kernel<1><<<dim3(HD / 64, BATCH * SEQ / 64), 32, 0, stream>>>(
        keyi, wkh, bk, kbuf, HD, 1.f);
    proj_in_kernel<0><<<dim3(HD / 64, BATCH * SEQ / 64), 32, 0, stream>>>(
        value, wvh, bv, vbuf, HD, 1.f);
    transpose_k_kernel<<<(BATCH * SEQ * HD + 255) / 256, 256, 0, stream>>>(
        kbuf, ktbuf);
    attn_kernel<<<dim3(SEQ / 64, NH, BATCH), 32, 0, stream>>>(
        qbuf, ktbuf, vbuf, xbuf);
    proj_out_kernel<<<dim3(CH / 64, BATCH * SEQ / 64), 32, 0, stream>>>(
        xbuf, wph, bp, out);
}
